// Adaptive_GConv_80659485819272
// MI455X (gfx1250) — matrix operand order repaired and checked by reading, not compiled, not run on hardware
//
#include <hip/hip_runtime.h>
#include <hip/hip_fp16.h>

typedef __attribute__((ext_vector_type(16))) _Float16 v16h;
typedef __attribute__((ext_vector_type(8)))  float    v8f;

#define NB   16
#define NC   256
#define NH   128
#define NW   128
#define KK   21
#define PAD  10
#define NHW  (NH * NW)

// LDS strides padded to an odd dword count (129 dwords = 258 halves) so that
// per-lane fragment gathers across rows/pixels land on distinct banks.
#define W1_STRIDE 258
#define X_STRIDE  258

// ---------------------------------------------------------------------------
// CDNA5 async global->LDS copy (one 32-bit element per lane, ASYNCcnt).
// dst: LDS byte offset (low 32 bits of the generic shared pointer),
// src: per-lane global address.
// ---------------------------------------------------------------------------
__device__ __forceinline__ void async_ld32(unsigned lds_off, const float* g) {
  asm volatile("global_load_async_to_lds_b32 %0, %1, off"
               :: "v"(lds_off), "v"(g) : "memory");
}
__device__ __forceinline__ void wait_async0() {
  asm volatile("s_wait_asynccnt 0" ::: "memory");
}

// ---------------------------------------------------------------------------
// Kernel 0: zero the per-batch prediction accumulators in workspace.
// ---------------------------------------------------------------------------
__global__ void agc_init_ws(float* __restrict__ wsSum) {
  if (threadIdx.x < NB) wsSum[threadIdx.x] = 0.0f;
}

// ---------------------------------------------------------------------------
// Kernel 1: kernel-size predictor.
//   h    = relu(W1[32,256] @ x[:,pix] + b1)        -> WMMA f32_16x16x32_f16
//   pred = relu(w2[32] . h + b2) * 20 + 1
//   wsSum[b] += sum(pred over pixels handled by this block)
// Block = 128 threads (4 waves). Each wave owns 16 pixels; block owns 64.
// ---------------------------------------------------------------------------
__global__ __launch_bounds__(128)
void agc_predict(const float* __restrict__ xg,
                 const float* __restrict__ w1,
                 const float* __restrict__ b1,
                 const float* __restrict__ w2,
                 const float* __restrict__ b2,
                 float* __restrict__ wsSum) {
  __shared__ _Float16 sW1[32 * W1_STRIDE];   // [out_ch][in_ch] f16
  __shared__ _Float16 sX [64 * X_STRIDE];    // [pixel][in_ch]  f16
  __shared__ float    sB1[32];
  __shared__ float    sW2[32];
  __shared__ float    sRed[128];

  const int tid  = threadIdx.x;
  const int b    = blockIdx.y;
  const int pix0 = blockIdx.x * 64;

  // Stage W1 (32x256) into LDS as f16.
  for (int i = tid; i < 32 * 256; i += 128) {
    int m = i >> 8, k = i & 255;
    sW1[m * W1_STRIDE + k] = (_Float16)w1[i];
  }
  if (tid < 32) { sB1[tid] = b1[tid]; sW2[tid] = w2[tid]; }

  // Stage 64 pixels x 256 channels of x into LDS as f16.
  // i = (c<<6)|p -> lanes sweep consecutive pixels of one channel: coalesced.
  const float* xb = xg + (size_t)b * NC * NHW;
  for (int i = tid; i < 64 * 256; i += 128) {
    int p = i & 63, c = i >> 6;
    sX[p * X_STRIDE + c] = (_Float16)xb[(size_t)c * NHW + pix0 + p];
  }
  __syncthreads();

  const int lane = tid & 31;
  const int wave = tid >> 5;          // 0..3 -> 16-pixel group
  const int lo   = lane & 15;
  const int hi   = lane >> 4;         // lane half (wave32)
  const int pixB = wave * 16;

  v8f c0 = {};   // out channels  0..15
  v8f c1 = {};   // out channels 16..31

  // K loop: 256 input channels in chunks of 32, fully unrolled -> 16 WMMAs.
  #pragma unroll
  for (int kk = 0; kk < 8; ++kk) {
    v16h a0{}, a1{}, bf{};
    // A fragment (16x32 f16): lane lo rows M=lo / M=lo+16; element e holds
    // K = (e&7) + (e>=8 ? 16 : 0) + hi*8 within the 32-wide chunk.
    #pragma unroll
    for (int e = 0; e < 16; ++e) {
      int k = kk * 32 + (e & 7) + ((e >> 3) << 4) + hi * 8;
      a0[e] = sW1[lo        * W1_STRIDE + k];
      a1[e] = sW1[(lo + 16) * W1_STRIDE + k];
    }
    // B fragment (32x16 f16): N = lo (pixel); element e holds
    // K = (e&7) + (e>=8 ? 16 : 0) + hi*8 within the chunk (same as A).
    #pragma unroll
    for (int e = 0; e < 16; ++e) {
      int k = kk * 32 + (e & 7) + ((e >> 3) << 4) + hi * 8;
      bf[e] = sX[(pixB + lo) * X_STRIDE + k];
    }
    c0 = __builtin_amdgcn_wmma_f32_16x16x32_f16(false, a0, false, bf,
                                                (short)0, c0, false, false);
    c1 = __builtin_amdgcn_wmma_f32_16x16x32_f16(false, a1, false, bf,
                                                (short)0, c1, false, false);
  }

  // C layout: VGPR r, lane l -> M = r + (l>=16 ? 8 : 0), N = l&15.
  // Per-lane partial of the w2 dot over its 16 resident rows.
  float partial = 0.0f;
  #pragma unroll
  for (int r = 0; r < 8; ++r) {
    int m0 = r + hi * 8;
    int m1 = 16 + r + hi * 8;
    float h0 = fmaxf(c0[r] + sB1[m0], 0.0f);
    float h1 = fmaxf(c1[r] + sB1[m1], 0.0f);
    partial = fmaf(sW2[m0], h0, partial);
    partial = fmaf(sW2[m1], h1, partial);
  }
  // Combine the two lane halves (same pixel N, disjoint M rows).
  partial += __shfl_xor(partial, 16, 32);

  float bias2 = b2[0];
  float pred  = fmaxf(partial + bias2, 0.0f) * 20.0f + 1.0f;
  sRed[tid] = (hi == 0) ? pred : 0.0f;   // count each pixel once
  __syncthreads();

  for (int s = 64; s > 0; s >>= 1) {
    if (tid < s) sRed[tid] += sRed[tid + s];
    __syncthreads();
  }
  if (tid == 0) atomicAdd(&wsSum[b], sRed[0]);
}

// ---------------------------------------------------------------------------
// Kernel 2: per-sample adaptive radius from mean prediction.
// ---------------------------------------------------------------------------
__global__ void agc_finalize(const float* __restrict__ wsSum,
                             int* __restrict__ wsRad) {
  int t = threadIdx.x;
  if (t < NB) {
    float mean = wsSum[t] * (1.0f / (float)NHW);
    float ksz  = fminf(fmaxf(floorf(mean), 1.0f), (float)KK);
    wsRad[t]   = (int)floorf((ksz - 1.0f) * 0.5f);
  }
}

// ---------------------------------------------------------------------------
// Kernel 3: depthwise conv with reflect-101 padding and per-sample radius.
// Block = 256 threads; 32x32 output tile per (b, c); 52x52 halo tile in LDS,
// filled with GLOBAL_LOAD_ASYNC_TO_LDS_B32 (reflect gather folded into the
// per-lane global address). Each thread computes a 1x4 output micro-tile with
// a sliding 4-register window over the horizontal taps: 1 ds_load + 4 FMAs
// per tap instead of 4+4. rad is uniform per block -> no divergence.
// ---------------------------------------------------------------------------
__global__ __launch_bounds__(256)
void agc_dwconv(const float* __restrict__ xg,
                const float* __restrict__ gk,
                const int* __restrict__ wsRad,
                float* __restrict__ out) {
  __shared__ float sTile[52 * 53];   // stride 53 (odd) -> bank friendly
  __shared__ float sKer[KK * KK];

  const int tid = threadIdx.x;
  const int b   = blockIdx.z;
  const int ch  = blockIdx.y;
  const int ty  = blockIdx.x >> 2, tx = blockIdx.x & 3;
  const int y0  = ty * 32, x0 = tx * 32;
  const int rad = wsRad[b];

  const float* xp = xg + (size_t)(b * NC + ch) * NHW;

  // gfx1250 prefetch of the next channel plane's tile (global_prefetch_b8).
  if (ch + 1 < NC)
    __builtin_prefetch(xp + NHW + (size_t)y0 * NW + x0, 0, 1);

  // Async halo-tile fill with reflect-101 indexing folded into the address.
  for (int i = tid; i < 52 * 52; i += 256) {
    int r = i / 52, c = i - r * 52;
    int gy = y0 + r - PAD;
    gy = gy < 0 ? -gy : (gy >= NH ? 2 * NH - 2 - gy : gy);
    int gx = x0 + c - PAD;
    gx = gx < 0 ? -gx : (gx >= NW ? 2 * NW - 2 - gx : gx);
    async_ld32((unsigned)(uintptr_t)&sTile[r * 53 + c], xp + gy * NW + gx);
  }
  // Async fill of this channel's 21x21 kernel.
  for (int i = tid; i < KK * KK; i += 256)
    async_ld32((unsigned)(uintptr_t)&sKer[i], gk + ch * (KK * KK) + i);

  wait_async0();
  __syncthreads();

  const int r = tid >> 3;        // output row within tile: 0..31
  const int g = tid & 7;         // column group: 4 consecutive outputs
  const int x = g * 4;
  float acc0 = 0.f, acc1 = 0.f, acc2 = 0.f, acc3 = 0.f;
  const int taps = 2 * rad + 1;

  for (int di = -rad; di <= rad; ++di) {
    const float* row  = &sTile[(r + PAD + di) * 53 + x + PAD - rad];
    const float* krow = &sKer[(di + PAD) * KK + (PAD - rad)];
    float w0 = row[0], w1 = row[1], w2 = row[2];
    for (int j = 0; j < taps; ++j) {
      float w3 = row[j + 3];
      float kw = krow[j];
      acc0 = fmaf(kw, w0, acc0);
      acc1 = fmaf(kw, w1, acc1);
      acc2 = fmaf(kw, w2, acc2);
      acc3 = fmaf(kw, w3, acc3);
      w0 = w1; w1 = w2; w2 = w3;
    }
  }

  float* ob = out + (size_t)(b * NC + ch) * NHW;
  float4 o4 = make_float4(acc0, acc1, acc2, acc3);
  *(float4*)&ob[(y0 + r) * NW + x0 + x] = o4;   // coalesced global_store_b128
}

// ---------------------------------------------------------------------------
extern "C" void kernel_launch(void* const* d_in, const int* in_sizes, int n_in,
                              void* d_out, int out_size, void* d_ws, size_t ws_size,
                              hipStream_t stream) {
  const float* x   = (const float*)d_in[0];   // [B,C,H,W]
  const float* gk  = (const float*)d_in[1];   // [C,1,K,K]
  const float* w1  = (const float*)d_in[2];   // [32,C,1,1]
  const float* b1v = (const float*)d_in[3];   // [32]
  const float* w2  = (const float*)d_in[4];   // [1,32,1,1]
  const float* b2v = (const float*)d_in[5];   // [1]
  float* out = (float*)d_out;

  float* wsSum = (float*)d_ws;                     // NB floats
  int*   wsRad = (int*)((char*)d_ws + 256);        // NB ints, aligned

  agc_init_ws<<<1, 32, 0, stream>>>(wsSum);

  dim3 gA(NHW / 64, NB);                           // 256 x 16 blocks
  agc_predict<<<gA, 128, 0, stream>>>(x, w1, b1v, w2, b2v, wsSum);

  agc_finalize<<<1, 32, 0, stream>>>(wsSum, wsRad);

  dim3 gC(16, NC, NB);                             // (4x4 tiles) x C x B
  agc_dwconv<<<gC, 256, 0, stream>>>(x, gk, wsRad, out);
}
